// FCVQVae_87316685128443
// MI455X (gfx1250) — matrix operand order repaired and checked by reading, not compiled, not run on hardware
//
#include <hip/hip_runtime.h>
#include <hip/hip_bf16.h>
#include <math.h>

// ---------------------------------------------------------------------------
// Types for CDNA5 WMMA (wave32): A/B = 16 bf16 per lane, C/D = 8 f32 per lane
// ---------------------------------------------------------------------------
typedef __attribute__((ext_vector_type(16))) __bf16        bf16x16;
typedef __attribute__((ext_vector_type(8)))  float         f32x8;
typedef __attribute__((ext_vector_type(8)))  unsigned short u16x8;

struct U16x16 { u16x8 lo, hi; };

__device__ __forceinline__ bf16x16 frag_cat(u16x8 lo, u16x8 hi) {
  U16x16 t; t.lo = lo; t.hi = hi;
  return __builtin_bit_cast(bf16x16, t);
}

// bf16 relu = zero out anything with the sign bit set (lowers to v_pk_max_i16)
__device__ __forceinline__ u16x8 relu_pk(u16x8 v) {
#pragma unroll
  for (int i = 0; i < 8; ++i) v[i] = (v[i] & 0x8000u) ? (unsigned short)0 : v[i];
  return v;
}

#define GF_RELUA  1   // relu applied to A on load
#define GF_RELUO  2   // relu applied to output
#define GF_RES    4   // add residual (bf16, ld = ldo)
#define GF_F32O   8   // store f32 output (ld = Nstore)
#define GF_BF16O  16  // store bf16 output (ld = ldo)
#define GF_NGUARD 32  // Nstore < Npad: guard column stores (out_proj2 only)

// ---------------------------------------------------------------------------
// Fused GEMM:  O = epilogue( act(A) @ Wt^T + bias [+ Res] )
//   A  : M x KPAD row-major bf16
//   Wt : Npad x KPAD row-major bf16 (pre-transposed weight)
// Block tile 128x64 (grid = Npad/64 x M/128), 8 waves, wave tile 32x32.
// KPAD and FLAGS are compile-time: branch-free inner loop, straight-line
// epilogue except when GF_NGUARD is set.
// ---------------------------------------------------------------------------
template<int KPAD, int FLAGS>
__global__ __launch_bounds__(256) void k_gemm(
    const __bf16* __restrict__ A, const __bf16* __restrict__ Wt,
    const float* __restrict__ bias, const __bf16* __restrict__ Res,
    __bf16* __restrict__ Obf, float* __restrict__ Of32,
    int Nstore, int ldo)
{
  const int lane  = threadIdx.x & 31;
  const int wave  = threadIdx.x >> 5;
  const int laneR = lane & 15;
  const int half  = lane >> 4;
  const int tileM = blockIdx.y * 128 + (wave & 3) * 32;
  const int tileN = blockIdx.x * 64  + (wave >> 2) * 32;

  const unsigned short* Au = (const unsigned short*)A;
  const unsigned short* Bu = (const unsigned short*)Wt;
  // A-fragment (16x32 bf16): lanes 0-15 hold K {0..7,16..23}, lanes 16-31 {8..15,24..31}
  const unsigned short* a0p = Au + (size_t)(tileM + laneR) * KPAD + half * 8;
  const unsigned short* a1p = a0p + (size_t)16 * KPAD;
  // B-fragment (32x16 bf16): lanes 0-15 hold K {0..7,16..23} of col n, lanes 16-31 {8..15,24..31}
  const unsigned short* b0p = Bu + (size_t)(tileN + laneR) * KPAD + half * 8;
  const unsigned short* b1p = b0p + (size_t)16 * KPAD;

  f32x8 acc00 = {}; f32x8 acc01 = {}; f32x8 acc10 = {}; f32x8 acc11 = {};

#pragma unroll 2
  for (int k0 = 0; k0 < KPAD; k0 += 32) {
    u16x8 a0l = *(const u16x8*)(a0p + k0);
    u16x8 a0h = *(const u16x8*)(a0p + k0 + 16);
    u16x8 a1l = *(const u16x8*)(a1p + k0);
    u16x8 a1h = *(const u16x8*)(a1p + k0 + 16);
    u16x8 b0l = *(const u16x8*)(b0p + k0);
    u16x8 b0h = *(const u16x8*)(b0p + k0 + 16);
    u16x8 b1l = *(const u16x8*)(b1p + k0);
    u16x8 b1h = *(const u16x8*)(b1p + k0 + 16);
    // speculative prefetch of the next K tile (overshoot on last iter is dropped)
    __builtin_prefetch(a0p + k0 + 32, 0, 1);
    __builtin_prefetch(b0p + k0 + 32, 0, 1);
    if constexpr ((FLAGS & GF_RELUA) != 0) {
      a0l = relu_pk(a0l); a0h = relu_pk(a0h);
      a1l = relu_pk(a1l); a1h = relu_pk(a1h);
    }
    bf16x16 fa0 = frag_cat(a0l, a0h), fa1 = frag_cat(a1l, a1h);
    bf16x16 fb0 = frag_cat(b0l, b0h), fb1 = frag_cat(b1l, b1h);
    acc00 = __builtin_amdgcn_wmma_f32_16x16x32_bf16(false, fa0, false, fb0, (short)0, acc00, false, false);
    acc01 = __builtin_amdgcn_wmma_f32_16x16x32_bf16(false, fa0, false, fb1, (short)0, acc01, false, false);
    acc10 = __builtin_amdgcn_wmma_f32_16x16x32_bf16(false, fa1, false, fb0, (short)0, acc10, false, false);
    acc11 = __builtin_amdgcn_wmma_f32_16x16x32_bf16(false, fa1, false, fb1, (short)0, acc11, false, false);
  }

  const int rowOff = half * 8;
  const int col0 = tileN + laneR, col1 = tileN + 16 + laneR;
  constexpr bool NG = (FLAGS & GF_NGUARD) != 0;

  // hoisted bias reads: one per column pair, not per element
  float bias0, bias1;
  if constexpr (NG) {
    bias0 = (col0 < Nstore) ? bias[col0] : 0.0f;
    bias1 = (col1 < Nstore) ? bias[col1] : 0.0f;
  } else {
    bias0 = bias[col0];
    bias1 = bias[col1];
  }

  auto emit = [&](int r, int c, float v, float bv) {
    if constexpr (NG) { if (c >= Nstore) return; }
    v += bv;
    if constexpr ((FLAGS & GF_RES) != 0)   v += (float)Res[(size_t)r * ldo + c];
    if constexpr ((FLAGS & GF_RELUO) != 0) v = fmaxf(v, 0.0f);
    if constexpr ((FLAGS & GF_BF16O) != 0) Obf[(size_t)r * ldo + c] = (__bf16)v;
    if constexpr ((FLAGS & GF_F32O) != 0)  Of32[(size_t)r * Nstore + c] = v;
  };
#pragma unroll
  for (int i = 0; i < 8; ++i) {
    int r0 = tileM + rowOff + i, r1 = r0 + 16;   // C/D layout: VGPR i -> M=i / i+8
    emit(r0, col0, acc00[i], bias0);
    emit(r0, col1, acc01[i], bias1);
    emit(r1, col0, acc10[i], bias0);
    emit(r1, col1, acc11[i], bias1);
  }
}

// ---------------------------------------------------------------------------
// Weight f32 (K x N) -> bf16 transposed (Npad x Kpad), zero-padded
// ---------------------------------------------------------------------------
__global__ void k_cvt_wt(const float* __restrict__ W, __bf16* __restrict__ Wt,
                         int K, int N, int Kpad, int Npad)
{
  int id = blockIdx.x * blockDim.x + threadIdx.x;
  int tot = Kpad * Npad;
  if (id >= tot) return;
  int n = id / Kpad, k = id % Kpad;
  float v = (k < K && n < N) ? W[(size_t)k * N + n] : 0.0f;
  Wt[id] = (__bf16)v;
}

// features f32 (4096 x 63) -> bf16 (4096 x 64) zero-padded
__global__ void k_cvt_feats(const float* __restrict__ F, __bf16* __restrict__ X)
{
  int id = blockIdx.x * blockDim.x + threadIdx.x;  // 4096*64 threads
  int m = id >> 6, k = id & 63;
  X[id] = (k < 63) ? (__bf16)F[(size_t)m * 63 + k] : (__bf16)0.0f;
}

__global__ void k_zero(float* p, int n) {
  int i = blockIdx.x * blockDim.x + threadIdx.x;
  if (i < n) p[i] = 0.0f;
}

// ---------------------------------------------------------------------------
// VQ: per row r (512 total, 64 threads): argmin over 512 codes, write xq,
// count histogram and commit-loss partial sums.
// ---------------------------------------------------------------------------
__global__ __launch_bounds__(64) void k_vq(
    const float* __restrict__ xe, const float* __restrict__ cb,
    int* __restrict__ outIdx, float* __restrict__ xq,
    float* __restrict__ counts, float* __restrict__ lossAcc)
{
  __shared__ float xs[64];
  __shared__ float bestD[64];
  __shared__ int   bestI[64];
  __shared__ float red[64];
  int r = blockIdx.x, t = threadIdx.x;
  xs[t] = xe[(size_t)r * 64 + t];
  __syncthreads();
  float bd = 3.4e38f; int bi = 0;
  for (int c = t; c < 512; c += 64) {
    const float* cr = cb + (size_t)c * 64;
    float d = 0.0f;
#pragma unroll 8
    for (int k = 0; k < 64; ++k) { float df = xs[k] - cr[k]; d += df * df; }
    if (d < bd) { bd = d; bi = c; }
  }
  bestD[t] = bd; bestI[t] = bi;
  __syncthreads();
  for (int s = 32; s > 0; s >>= 1) {
    if (t < s && bestD[t + s] < bestD[t]) { bestD[t] = bestD[t + s]; bestI[t] = bestI[t + s]; }
    __syncthreads();
  }
  int best = bestI[0];
  float q = cb[(size_t)best * 64 + t];
  xq[(size_t)r * 64 + t] = q;
  float df = xs[t] - q;
  red[t] = df * df;
  __syncthreads();
  for (int s = 32; s > 0; s >>= 1) { if (t < s) red[t] += red[t + s]; __syncthreads(); }
  if (t == 0) {
    outIdx[r] = best;
    atomicAdd(&counts[best], 1.0f);
    atomicAdd(lossAcc, red[0]);
  }
}

// scal[b*64+l] = ||xq[b,l,:]|| / ||xe[b,l,:]||   (norms over D)
__global__ __launch_bounds__(64) void k_scal(
    const float* __restrict__ xe, const float* __restrict__ xq, float* __restrict__ scal)
{
  __shared__ float se[64], sq[64];
  int r = blockIdx.x, t = threadIdx.x;
  float e = xe[(size_t)r * 64 + t], q = xq[(size_t)r * 64 + t];
  se[t] = e * e; sq[t] = q * q;
  __syncthreads();
  for (int s = 32; s > 0; s >>= 1) { if (t < s) { se[t] += se[t + s]; sq[t] += sq[t + s]; } __syncthreads(); }
  if (t == 0) scal[r] = sqrtf(sq[0]) / sqrtf(se[0]);
}

__device__ __forceinline__ float blockReduce64(float v, float* red, int t) {
  __syncthreads();
  red[t] = v;
  __syncthreads();
  for (int s = 32; s > 0; s >>= 1) { if (t < s) red[t] += red[t + s]; __syncthreads(); }
  return red[0];
}

// rotation trick, rank-1 form: block per (b,d), threads over l
// out[b,d,l] = e - 2 r (r.e) + 2 qhat (ehat.e);  decin[b,l,d] = scal[b,l]*out
__global__ __launch_bounds__(64) void k_rot(
    const float* __restrict__ xe, const float* __restrict__ xq,
    const float* __restrict__ scal, __bf16* __restrict__ decin)
{
  __shared__ float red[64];
  int bd = blockIdx.x, b = bd >> 6, d = bd & 63;
  int l = threadIdx.x;
  size_t idx = ((size_t)(b * 64 + l)) * 64 + d;
  float e = xe[idx], q = xq[idx];
  float se2 = blockReduce64(e * e, red, l);
  float sq2 = blockReduce64(q * q, red, l);
  float ne = sqrtf(se2), nq = sqrtf(sq2);
  float neC = fmaxf(ne, 1e-12f), nqC = fmaxf(nq, 1e-12f);
  float eh = e / neC, qh = q / nqC;
  float s = eh + qh;
  float ss2 = blockReduce64(s * s, red, l);
  float nsC = fmaxf(sqrtf(ss2), 1e-12f);
  float rr = s / nsC;
  float rd = blockReduce64(rr * e, red, l);    // r . e
  float dhe = se2 / neC;                        // ehat . e
  float out = e - 2.0f * rd * rr + 2.0f * dhe * qh;
  decin[idx] = (__bf16)(scal[b * 64 + l] * out);
}

// decoder un-concat: in (B*l, 1024) -> out (B*2l, 512)
// out[b, j*l+i, c] = in[b*l+i, j*512+c]
__global__ void k_shuffle(const __bf16* __restrict__ in, __bf16* __restrict__ out, int l, int total)
{
  int id = blockIdx.x * blockDim.x + threadIdx.x;
  if (id >= total) return;
  int c = id & 1023;
  int row = id >> 10;
  int i = row % l, b = row / l;
  int j = c >> 9, cc = c & 511;
  out[((size_t)(b * 2 * l + j * l + i)) * 512 + cc] = in[id];
}

// commit loss mean + perplexity -> d_out tail
__global__ __launch_bounds__(512) void k_stats(
    const float* __restrict__ counts, const float* __restrict__ lossAcc, float* __restrict__ out2)
{
  __shared__ float red[512];
  int t = threadIdx.x;
  float p = counts[t] / 512.0f;
  red[t] = p * logf(p + 1e-7f);
  __syncthreads();
  for (int s = 256; s > 0; s >>= 1) { if (t < s) red[t] += red[t + s]; __syncthreads(); }
  if (t == 0) {
    out2[0] = lossAcc[0] / (512.0f * 64.0f);
    out2[1] = expf(-red[0]);
  }
}

// ---------------------------------------------------------------------------
// Host orchestration
// ---------------------------------------------------------------------------
struct WD { int idx, K, N; };

static void launch_gemm(const __bf16* A, const __bf16* Wt, const float* bias,
                        const __bf16* Res, __bf16* Obf, float* Of32,
                        int M, int N, int Np, int Kp, int flags, hipStream_t stream)
{
  dim3 g(Np / 64, M / 128), blk(256);
#define GCASE(KP, FL)                                                         \
  if (Kp == (KP) && flags == (FL)) {                                          \
    k_gemm<(KP), (FL)><<<g, blk, 0, stream>>>(A, Wt, bias, Res, Obf, Of32, N, N); \
    return;                                                                   \
  }
  GCASE(64,   GF_RELUO | GF_BF16O)            // enc/dec in_proj
  GCASE(512,  GF_RELUA | GF_BF16O)            // fc1
  GCASE(512,  GF_BF16O)                       // dec lin
  GCASE(512,  GF_F32O)                        // out_proj (N=64 exact)
  GCASE(512,  GF_F32O | GF_NGUARD)            // out_proj2 (N=63 < Npad=64)
  GCASE(512,  GF_RELUO | GF_BF16O)            // out_proj1
  GCASE(1024, GF_BF16O)                       // enc lin
  GCASE(1024, GF_RELUA | GF_RES | GF_BF16O)   // fc2
#undef GCASE
}

extern "C" void kernel_launch(void* const* d_in, const int* in_sizes, int n_in,
                              void* d_out, int out_size, void* d_ws, size_t ws_size,
                              hipStream_t stream) {
  (void)in_sizes; (void)n_in; (void)out_size; (void)ws_size;

  // ---- weight descriptor table (setup_inputs dict order, [W,b] pairs) ----
  WD wds[47]; int nw = 0; int p = 1;
  wds[nw++] = {p, 63, 512}; p += 2;                    // enc.in_proj
  for (int L = 0; L < 3; ++L) {                        // enc.down
    wds[nw++] = {p, 1024, 512}; p += 2;                // lin
    for (int b = 0; b < 3; ++b) {
      wds[nw++] = {p, 512, 1024}; p += 2;              // fc1
      wds[nw++] = {p, 1024, 512}; p += 2;              // fc2
    }
  }
  wds[nw++] = {p, 512, 64}; p += 2;                    // enc.out_proj (p=45)
  wds[nw++] = {p, 64, 512}; p += 2;                    // dec.in_proj (p=47)
  for (int L = 0; L < 3; ++L) {                        // dec.up: blocks then lin
    for (int b = 0; b < 3; ++b) {
      wds[nw++] = {p, 512, 1024}; p += 2;
      wds[nw++] = {p, 1024, 512}; p += 2;
    }
    wds[nw++] = {p, 512, 1024}; p += 2;                // lin
  }
  wds[nw++] = {p, 512, 512}; p += 2;                   // out_proj1 (p=91)
  wds[nw++] = {p, 512, 63};  p += 2;                   // out_proj2 (p=93)
  const float* CB = (const float*)d_in[p];             // codebook (p=95)
  const float* F  = (const float*)d_in[0];

  // ---- workspace carve ----
  char* w = (char*)d_ws;
  size_t off = 0;
  auto carve = [&](size_t bytes) -> void* {
    void* q = w + off;
    off += (bytes + 255) & ~(size_t)255;
    return q;
  };
  __bf16* wt[47];
  for (int i = 0; i < nw; ++i) {
    int Kp = (wds[i].K + 31) & ~31, Np = (wds[i].N + 63) & ~63;
    wt[i] = (__bf16*)carve((size_t)Kp * Np * 2);
  }
  const size_t BUFSZ = (size_t)4 << 20;                // 4 MB, max act = 2048x1024 bf16
  __bf16* bufs[3] = { (__bf16*)carve(BUFSZ), (__bf16*)carve(BUFSZ), (__bf16*)carve(BUFSZ) };
  float* xe     = (float*)carve(512 * 64 * 4);
  float* xq     = (float*)carve(512 * 64 * 4);
  float* scal   = (float*)carve(512 * 4);
  int*   vidx   = (int*)carve(512 * 4);
  float* counts = (float*)carve(513 * 4);              // [512] = loss accumulator

  // ---- convert all weights once (transposed, padded, bf16) ----
  for (int i = 0; i < nw; ++i) {
    int Kp = (wds[i].K + 31) & ~31, Np = (wds[i].N + 63) & ~63;
    int tot = Kp * Np;
    k_cvt_wt<<<dim3((tot + 255) / 256), dim3(256), 0, stream>>>(
        (const float*)d_in[wds[i].idx], wt[i], wds[i].K, wds[i].N, Kp, Np);
  }

  auto gemm = [&](const __bf16* A, int wi, const __bf16* Res,
                  __bf16* Obf, float* Of32, int M, int flags) {
    int K = wds[wi].K, N = wds[wi].N;
    int Kp = (K + 31) & ~31, Np = (N + 63) & ~63;
    const float* bias = (const float*)d_in[wds[wi].idx + 1];
    launch_gemm(A, wt[wi], bias, Res, Obf, Of32, M, N, Np, Kp, flags, stream);
  };

  // ---- encoder ----
  k_cvt_feats<<<dim3((4096 * 64) / 256), dim3(256), 0, stream>>>(F, bufs[0]);
  int xi = 0, wi = 0;
  gemm(bufs[0], wi++, nullptr, bufs[1], nullptr, 4096, GF_RELUO | GF_BF16O);  // in_proj
  xi = 1;
  int M = 2048;                                        // (B*T/2) rows after free reshape
  for (int L = 0; L < 3; ++L) {
    int oi = (xi + 1) % 3;
    gemm(bufs[xi], wi++, nullptr, bufs[oi], nullptr, M, GF_BF16O);            // lin (2H->H)
    xi = oi;
    for (int b = 0; b < 3; ++b) {
      int hi = (xi + 1) % 3, zi = (xi + 2) % 3;
      gemm(bufs[xi], wi++, nullptr, bufs[hi], nullptr, M, GF_RELUA | GF_BF16O);          // fc1
      gemm(bufs[hi], wi++, bufs[xi], bufs[zi], nullptr, M, GF_RELUA | GF_RES | GF_BF16O); // fc2
      xi = zi;
    }
    M >>= 1;
  }
  gemm(bufs[xi], wi++, nullptr, nullptr, xe, 512, GF_F32O);                    // out_proj -> f32

  // ---- VQ + rotation trick ----
  k_zero<<<dim3(3), dim3(256), 0, stream>>>(counts, 513);
  k_vq<<<dim3(512), dim3(64), 0, stream>>>(xe, CB, vidx, xq, counts, counts + 512);
  k_scal<<<dim3(512), dim3(64), 0, stream>>>(xe, xq, scal);
  k_rot<<<dim3(512), dim3(64), 0, stream>>>(xe, xq, scal, bufs[0]);            // decin bf16
  xi = 0;

  // ---- decoder ----
  gemm(bufs[0], wi++, nullptr, bufs[1], nullptr, 512, GF_RELUO | GF_BF16O);    // dec.in_proj
  xi = 1;
  M = 512;
  for (int L = 0; L < 3; ++L) {
    for (int b = 0; b < 3; ++b) {
      int hi = (xi + 1) % 3, zi = (xi + 2) % 3;
      gemm(bufs[xi], wi++, nullptr, bufs[hi], nullptr, M, GF_RELUA | GF_BF16O);
      gemm(bufs[hi], wi++, bufs[xi], bufs[zi], nullptr, M, GF_RELUA | GF_RES | GF_BF16O);
      xi = zi;
    }
    int ri = (xi + 1) % 3, si = (xi + 2) % 3;
    gemm(bufs[xi], wi++, nullptr, bufs[ri], nullptr, M, GF_BF16O);             // lin H->2H
    int lcur = M / 8;                                                          // l per batch
    int total = M * 1024;
    k_shuffle<<<dim3((total + 255) / 256), dim3(256), 0, stream>>>(bufs[ri], bufs[si], lcur, total);
    xi = si; M <<= 1;
  }
  // M = 4096
  {
    int oi = (xi + 1) % 3;
    gemm(bufs[xi], wi++, nullptr, bufs[oi], nullptr, 4096, GF_RELUO | GF_BF16O); // out_proj1
    xi = oi;
  }
  gemm(bufs[xi], wi++, nullptr, nullptr, (float*)d_out, 4096, GF_F32O | GF_NGUARD); // out_proj2 -> d_out (N=63)

  // ---- scalars ----
  k_stats<<<dim3(1), dim3(512), 0, stream>>>(counts, counts + 512, (float*)d_out + 258048);
}
